// SpikingQueryKeyValue_87454124081923
// MI455X (gfx1250) — matrix operand order repaired and checked by reading, not compiled, not run on hardware
//
#include <hip/hip_runtime.h>

// ---------------------------------------------------------------------------
// Problem constants (from the reference)
// ---------------------------------------------------------------------------
#define BATCH   4
#define SEQ     4096
#define TOKENS  (BATCH * SEQ)      // 16384 rows (M)
#define DMODEL  1024               // reduction dim (K of the GEMM)
#define NCH     1024               // H*DHEAD = 16*64 output channels (N)
#define V_TH    1.0f
#define BETA    0.9048374180359595f  // exp(-DT/TAU) = exp(-0.1)

// LDS layout for the staged W strip: smem[d][n], d = 0..1023, n = 0..63,
// padded to stride 72 bf16 (144 B = 9*16 B: keeps b128 alignment, spreads banks)
#define LDS_STRIDE 72
#define LDS_BYTES  (DMODEL * LDS_STRIDE * 2)   // 147456 B <= 320 KB/WGP

typedef __attribute__((ext_vector_type(16))) __bf16 v16bf;
typedef __attribute__((ext_vector_type(8)))  __bf16 v8bf;
typedef __attribute__((ext_vector_type(8)))  float  v8f;

// ---------------------------------------------------------------------------
// A-fragment: 16x32 bf16 tile of X.
// ISA layout (16-bit A 16x32): lane L holds row M = L&15; K-base = 8*(L>>4);
// vector elems 0..7 -> K = kb+0..7, elems 8..15 -> K = kb+16..23.
// Two pairs of b128 loads per lane, converted f32->bf16 (v_cvt_pk_bf16_f32).
// ---------------------------------------------------------------------------
__device__ __forceinline__ v16bf load_a_frag(const float* __restrict__ x,
                                             int mrow, int lane, int k0) {
  const int row = mrow + (lane & 15);
  const int kb  = k0 + ((lane >> 4) << 3);
  const float* pr = x + (size_t)row * DMODEL + kb;
  float4 f0 = *(const float4*)(pr + 0);
  float4 f1 = *(const float4*)(pr + 4);
  float4 f2 = *(const float4*)(pr + 16);
  float4 f3 = *(const float4*)(pr + 20);
  v16bf a;
  a[0]  = (__bf16)f0.x; a[1]  = (__bf16)f0.y; a[2]  = (__bf16)f0.z; a[3]  = (__bf16)f0.w;
  a[4]  = (__bf16)f1.x; a[5]  = (__bf16)f1.y; a[6]  = (__bf16)f1.z; a[7]  = (__bf16)f1.w;
  a[8]  = (__bf16)f2.x; a[9]  = (__bf16)f2.y; a[10] = (__bf16)f2.z; a[11] = (__bf16)f2.w;
  a[12] = (__bf16)f3.x; a[13] = (__bf16)f3.y; a[14] = (__bf16)f3.z; a[15] = (__bf16)f3.w;
  return a;
}

// ---------------------------------------------------------------------------
// B-fragment from LDS: B[k][n] = W[nbase+n][k0+k] = smem[k0+k][n].
// ISA layout (16-bit B): lane L holds column N = L&15; K-base = 8*(L>>4);
// elems 0..7 -> K = kb+0..7, elems 8..15 -> K = kb+16..23 (strided LDS reads).
// ---------------------------------------------------------------------------
__device__ __forceinline__ v16bf load_b_lds(const __bf16* smem,
                                            int nsub, int lane, int k0) {
  const int d = k0 + ((lane >> 4) << 3);
  const __bf16* p = smem + (size_t)d * LDS_STRIDE + nsub * 16 + (lane & 15);
  v16bf b;
#pragma unroll
  for (int e = 0; e < 8; ++e) { b[e] = p[e * LDS_STRIDE]; b[8 + e] = p[(16 + e) * LDS_STRIDE]; }
  return b;
}

// ---------------------------------------------------------------------------
// GEMM: out[t][m][n] = sum_d x[m][d] * W_t[n][d]   (t = Q,K,V via blockIdx.z)
// Block = 256 threads = 8 waves. The block's 64-column W strip (64x1024 f32)
// is staged ONCE into LDS as transposed bf16; the k-loop reads B only from
// LDS. Each wave computes a 32(M) x 64(N) tile: 2 A-frags x 4 B-frags ->
// 8 v_wmma_f32_16x16x32_bf16 per 32-wide K chunk.
// Grid: (TOKENS/(32*8)=64, NCH/64=16, 3).
// ---------------------------------------------------------------------------
__global__ void __launch_bounds__(256)
sqkv_gemm_kernel(const float* __restrict__ x,
                 const float* __restrict__ Wq,
                 const float* __restrict__ Wk,
                 const float* __restrict__ Wv,
                 float* __restrict__ out) {
  extern __shared__ __bf16 smem[];   // [DMODEL][LDS_STRIDE]

  const int lane = threadIdx.x & 31;
  const int wave = threadIdx.x >> 5;

  const float* W = (blockIdx.z == 0) ? Wq : ((blockIdx.z == 1) ? Wk : Wv);
  float* o = out + (size_t)blockIdx.z * ((size_t)TOKENS * NCH);

  const int m_base = (blockIdx.x * 8 + wave) * 32;  // 32 rows per wave
  const int n_base = blockIdx.y * 64;               // 64 cols per wave

  // ---- Stage W strip: smem[d][n] = bf16(W[n_base+n][d]) -------------------
  // 64 rows x 1024 d = 16384 float4 loads across 256 threads (64 iters).
  // Global reads fully coalesced (consecutive tid -> consecutive d).
#pragma unroll 4
  for (int it = 0; it < 64; ++it) {
    const int idx = threadIdx.x + it * 256;
    const int n   = idx >> 8;             // 0..63
    const int d   = (idx & 255) << 2;     // 0..1020, step 4
    float4 w = *(const float4*)(W + (size_t)(n_base + n) * DMODEL + d);
    __bf16* q = smem + (size_t)d * LDS_STRIDE + n;
    q[0 * LDS_STRIDE] = (__bf16)w.x;
    q[1 * LDS_STRIDE] = (__bf16)w.y;
    q[2 * LDS_STRIDE] = (__bf16)w.z;
    q[3 * LDS_STRIDE] = (__bf16)w.w;
  }
  __syncthreads();

  // ---- Main K loop --------------------------------------------------------
  v8f acc[2][4];
#pragma unroll
  for (int i = 0; i < 2; ++i)
#pragma unroll
    for (int j = 0; j < 4; ++j)
      acc[i][j] = (v8f){0.f, 0.f, 0.f, 0.f, 0.f, 0.f, 0.f, 0.f};

  for (int k0 = 0; k0 < DMODEL; k0 += 32) {
    v16bf a0 = load_a_frag(x, m_base,      lane, k0);
    v16bf a1 = load_a_frag(x, m_base + 16, lane, k0);
    v16bf b[4];
#pragma unroll
    for (int j = 0; j < 4; ++j)
      b[j] = load_b_lds(smem, j, lane, k0);

#pragma unroll
    for (int j = 0; j < 4; ++j) {
      acc[0][j] = __builtin_amdgcn_wmma_f32_16x16x32_bf16(
          false, a0, false, b[j], (short)0, acc[0][j], false, false);
      acc[1][j] = __builtin_amdgcn_wmma_f32_16x16x32_bf16(
          false, a1, false, b[j], (short)0, acc[1][j], false, false);
    }
  }

  // ---- Store: C/D layout: vec elem r, lane L -> row = r + 8*(L>>4),
  // col = L&15. Lanes 0-15 / 16-31 each write one contiguous 64 B segment.
  const int lhalf = lane >> 4;
  const int lcol  = lane & 15;
#pragma unroll
  for (int i = 0; i < 2; ++i) {
#pragma unroll
    for (int j = 0; j < 4; ++j) {
#pragma unroll
      for (int r = 0; r < 8; ++r) {
        const int row = m_base + i * 16 + r + 8 * lhalf;
        const int col = n_base + j * 16 + lcol;
        o[(size_t)row * NCH + col] = acc[i][j][r];
      }
    }
  }
}

// ---------------------------------------------------------------------------
// LIF scan, in place over the GEMM output. One thread per (tensor, channel):
// 3 * 1024 = 3072 threads. Serial 16384-step recurrence per channel; the
// next 32-token group is double-buffered in registers (32 independent loads
// in flight against the serial fma->cmp->cndmask chain) and the stream is
// prefetched 512 tokens ahead via global_prefetch_b8. Reads/writes are
// 128 B-coalesced per wave (32 adjacent channels share each token row).
// ---------------------------------------------------------------------------
#define LIF_GRP 32

__global__ void __launch_bounds__(256)
sqkv_lif_kernel(float* __restrict__ qkv) {
  const int tensor = blockIdx.y;
  const int ch     = blockIdx.x * blockDim.x + threadIdx.x;
  float* p = qkv + (size_t)tensor * ((size_t)TOKENS * NCH) + ch;

  float v = 0.0f;
  float buf[LIF_GRP];
  float nbuf[LIF_GRP];

#pragma unroll
  for (int j = 0; j < LIF_GRP; ++j)
    buf[j] = p[(size_t)j * NCH];

  for (int t0 = 0; t0 < TOKENS; t0 += LIF_GRP) {
    const int t1 = t0 + LIF_GRP;

    // Deep prefetch: 16 groups (512 tokens) ahead -> global_prefetch_b8.
    const int tp = t0 + 16 * LIF_GRP;
    if (tp < TOKENS)
      __builtin_prefetch(p + (size_t)tp * NCH, 0, 0);

    if (t1 < TOKENS) {
#pragma unroll
      for (int j = 0; j < LIF_GRP; ++j)
        nbuf[j] = p[(size_t)(t1 + j) * NCH];
    }

#pragma unroll
    for (int j = 0; j < LIF_GRP; ++j) {
      const float vn    = v * BETA + buf[j];
      const bool  fired = (vn >= V_TH);
      p[(size_t)(t0 + j) * NCH] = fired ? 1.0f : 0.0f;
      v = fired ? 0.0f : vn;
    }

#pragma unroll
    for (int j = 0; j < LIF_GRP; ++j)
      buf[j] = nbuf[j];
  }
}

// ---------------------------------------------------------------------------
// Launch
// ---------------------------------------------------------------------------
extern "C" void kernel_launch(void* const* d_in, const int* in_sizes, int n_in,
                              void* d_out, int out_size, void* d_ws, size_t ws_size,
                              hipStream_t stream) {
  const float* x  = (const float*)d_in[0];
  const float* Wq = (const float*)d_in[1];
  const float* Wk = (const float*)d_in[2];
  const float* Wv = (const float*)d_in[3];
  float* out = (float*)d_out;

  // GEMM: 64 M-blocks (8 waves * 32 rows) x 16 N-strips x 3 tensors.
  // Dynamic LDS: 144 KB staged W strip (gfx1250 WGP has 320 KB).
  dim3 ggrid(TOKENS / (32 * 8), NCH / 64, 3);
  dim3 gblk(256);
  sqkv_gemm_kernel<<<ggrid, gblk, LDS_BYTES, stream>>>(x, Wq, Wk, Wv, out);

  // LIF scan: 1024 channels x 3 tensors, in place on d_out.
  dim3 lgrid(NCH / 256, 3);
  dim3 lblk(256);
  sqkv_lif_kernel<<<lgrid, lblk, 0, stream>>>(out);
}
